// DenseGATConv_84258668413540
// MI455X (gfx1250) — compile-verified, matrix operand order checked by reading, not run on hardware
//
#include <hip/hip_runtime.h>
#include <hip/hip_bf16.h>

typedef __attribute__((ext_vector_type(16))) _Float16 v16h;
typedef __attribute__((ext_vector_type(8)))  _Float16 v8h;
typedef __attribute__((ext_vector_type(8)))  float    v8f;

#define B_  4
#define N_  2048
#define F_  128
#define H_  4
#define C_  64
#define HC_ 256
#define NEG_SLOPE 0.2f

// ---------------------------------------------------------------------------
// Kernel 1: W [F,HC] f32  ->  Wt [HC,F] f16 (transposed + converted)
// ---------------------------------------------------------------------------
__global__ __launch_bounds__(256) void prep_w_kernel(
    const float* __restrict__ W, _Float16* __restrict__ Wt)
{
    int t = blockIdx.x * 256 + threadIdx.x;         // 0 .. F*HC-1
    if (t >= F_ * HC_) return;
    int k = t & (F_ - 1);                           // 0..127
    int n = t >> 7;                                 // 0..255
    Wt[(size_t)n * F_ + k] = (_Float16)W[(size_t)k * HC_ + n];
}

// ---------------------------------------------------------------------------
// Kernel 2: fused projection GEMM (WMMA f16) + attention coefficients.
//   xh = x @ W  (M = B*N = 8192, K = 128, N = 256)
// One block = 128 threads = 4 waves; wave w handles head h = w.
// Each wave: 16 rows x 64 cols (4 C-tiles), K-loop of 4 x 32.
// Outputs: xh_t f16 [B][H][C][N], a_src/a_dst f32 [B][H][N].
// ---------------------------------------------------------------------------
__global__ __launch_bounds__(128) void proj_kernel(
    const float*    __restrict__ x,
    const float*    __restrict__ att_src,
    const float*    __restrict__ att_dst,
    const _Float16* __restrict__ Wt,
    _Float16*       __restrict__ xh_t,
    float*          __restrict__ a_src,
    float*          __restrict__ a_dst)
{
    const int lane  = threadIdx.x & 31;
    const int h     = threadIdx.x >> 5;             // head = wave id (0..3)
    const int m0g   = blockIdx.x * 16;              // global row into B*N
    const int b     = m0g >> 11;                    // N_ = 2048
    const int n0    = m0g & (N_ - 1);
    const int ml    = lane & 15;
    const int khalf = (lane >> 4) << 3;             // 0 or 8 (WMMA A/B K layout)
    const int rbase = (lane < 16) ? 0 : 8;          // C-tile row base per half

    const float* xrow = x + ((size_t)(b * N_ + n0 + ml)) * F_;

    v8f acc[4] = {};
    for (int kb = 0; kb < F_; kb += 32) {
        // A tile: rows of x, converted to f16 in WMMA A layout
        v16h va;
        {
            const float4 f0 = *(const float4*)(xrow + kb + khalf);
            const float4 f1 = *(const float4*)(xrow + kb + khalf + 4);
            const float4 f2 = *(const float4*)(xrow + kb + 16 + khalf);
            const float4 f3 = *(const float4*)(xrow + kb + 16 + khalf + 4);
            va[0]  = (_Float16)f0.x; va[1]  = (_Float16)f0.y;
            va[2]  = (_Float16)f0.z; va[3]  = (_Float16)f0.w;
            va[4]  = (_Float16)f1.x; va[5]  = (_Float16)f1.y;
            va[6]  = (_Float16)f1.z; va[7]  = (_Float16)f1.w;
            va[8]  = (_Float16)f2.x; va[9]  = (_Float16)f2.y;
            va[10] = (_Float16)f2.z; va[11] = (_Float16)f2.w;
            va[12] = (_Float16)f3.x; va[13] = (_Float16)f3.y;
            va[14] = (_Float16)f3.z; va[15] = (_Float16)f3.w;
        }
        #pragma unroll
        for (int t = 0; t < 4; ++t) {
            const _Float16* wrow = Wt + (size_t)(h * C_ + t * 16 + ml) * F_;
            v8h lo = *(const v8h*)(wrow + kb + khalf);
            v8h hi = *(const v8h*)(wrow + kb + 16 + khalf);
            v16h vb;
            #pragma unroll
            for (int e = 0; e < 8; ++e) { vb[e] = lo[e]; vb[e + 8] = hi[e]; }
            acc[t] = __builtin_amdgcn_wmma_f32_16x16x32_f16(
                false, va, false, vb, (short)0, acc[t], false, false);
        }
    }

    // Fused attention coefficients: a_src/a_dst[row] = sum_c xh[row,c]*att[h,c]
    float psrc[8] = {0.f,0.f,0.f,0.f,0.f,0.f,0.f,0.f};
    float pdst[8] = {0.f,0.f,0.f,0.f,0.f,0.f,0.f,0.f};
    #pragma unroll
    for (int t = 0; t < 4; ++t) {
        int   c  = t * 16 + ml;
        float as = att_src[h * C_ + c];
        float ad = att_dst[h * C_ + c];
        #pragma unroll
        for (int r = 0; r < 8; ++r) {
            psrc[r] += acc[t][r] * as;
            pdst[r] += acc[t][r] * ad;
        }
    }
    #pragma unroll
    for (int m = 1; m < 16; m <<= 1) {
        #pragma unroll
        for (int r = 0; r < 8; ++r) {
            psrc[r] += __shfl_xor(psrc[r], m, 32);
            pdst[r] += __shfl_xor(pdst[r], m, 32);
        }
    }
    if (ml == 0) {
        float* srow = a_src + (size_t)(b * H_ + h) * N_ + n0 + rbase;
        float* drow = a_dst + (size_t)(b * H_ + h) * N_ + n0 + rbase;
        #pragma unroll
        for (int r = 0; r < 8; ++r) { srow[r] = psrc[r]; drow[r] = pdst[r]; }
    }

    // Store xh transposed as f16 [B][H][C][N] (contiguous in N for kernel 4)
    #pragma unroll
    for (int t = 0; t < 4; ++t) {
        int c = t * 16 + ml;
        _Float16* col = xh_t + ((size_t)((b * H_ + h) * C_ + c)) * N_ + n0 + rbase;
        #pragma unroll
        for (int r = 0; r < 8; ++r) col[r] = (_Float16)acc[t][r];
    }
}

// ---------------------------------------------------------------------------
// Kernel 3: per-(b,h) global max of a_src (softmax upper bound), 16 scalars.
// ---------------------------------------------------------------------------
__global__ __launch_bounds__(256) void amax_kernel(
    const float* __restrict__ a_src, float* __restrict__ amax)
{
    const int bh = blockIdx.x;                      // 0..15
    const float* p = a_src + (size_t)bh * N_;
    float m = -1e30f;
    for (int j = threadIdx.x; j < N_; j += 256) m = fmaxf(m, p[j]);
    #pragma unroll
    for (int s = 1; s < 32; s <<= 1) m = fmaxf(m, __shfl_xor(m, s, 32));
    __shared__ float sm[8];
    if ((threadIdx.x & 31) == 0) sm[threadIdx.x >> 5] = m;
    __syncthreads();
    if (threadIdx.x < 8) {
        float v = sm[threadIdx.x];
        #pragma unroll
        for (int s = 1; s < 8; s <<= 1) v = fmaxf(v, __shfl_xor(v, s, 32));
        if (threadIdx.x == 0) amax[bh] = v;
    }
}

// ---------------------------------------------------------------------------
// Kernel 4: fused masked softmax + aggregation, single pass.
// bound_i = leaky_relu(max_j a_src[j] + a_dst[i]) >= every row-i score, so
// exp(alpha - bound) never overflows and no online rescaling is needed.
// One wave handles TWO 16-row A-tiles (32 rows) of one (b,h): the 4 B-tile
// loads per j-chunk feed 8 WMMAs, halving V-panel L2 traffic.
// ---------------------------------------------------------------------------
__global__ __launch_bounds__(128) void gat_attn_kernel(
    const float*    __restrict__ adj,
    const float*    __restrict__ a_src_arr,
    const float*    __restrict__ a_dst_arr,
    const float*    __restrict__ amax_arr,
    const _Float16* __restrict__ xh_t,
    const float*    __restrict__ bias,
    float*          __restrict__ out)
{
    const int lane  = threadIdx.x & 31;
    const int wave  = threadIdx.x >> 5;
    const int tile  = blockIdx.x * 4 + wave;        // 0 .. B*H*(N/32)-1 = 1023
    const int it    = tile & 63;                    // 32-row tile within (b,h)
    const int bh    = tile >> 6;                    // 0..15
    const int h     = bh & (H_ - 1);
    const int b     = bh >> 2;
    const int i0    = it * 32;
    const int ml    = lane & 15;
    const int khalf = (lane >> 4) << 3;             // 0 or 8
    const int rbase = (lane < 16) ? 0 : 8;

    const float*    asrc  = a_src_arr + (size_t)bh * N_;
    const _Float16* vbase = xh_t + (size_t)bh * C_ * N_;
    const float     maxA  = amax_arr[bh];

    // two row groups: rows i0+ml and i0+16+ml
    int          iR[2];
    float        adstv[2], boundv[2];
    const float* adjrowv[2];
    #pragma unroll
    for (int g = 0; g < 2; ++g) {
        iR[g]      = i0 + g * 16 + ml;
        adstv[g]   = a_dst_arr[(size_t)bh * N_ + iR[g]];
        float bb   = maxA + adstv[g];
        boundv[g]  = (bb > 0.f) ? bb : NEG_SLOPE * bb;
        adjrowv[g] = adj + ((size_t)b * N_ + iR[g]) * N_;
    }

    float lsum[2] = {0.f, 0.f};
    v8f acc[2][4] = {};

    for (int j0 = 0; j0 < N_; j0 += 32) {
        const int jA = j0 + khalf;
        const int jB = j0 + 16 + khalf;

        const float4 s0 = *(const float4*)(asrc + jA);
        const float4 s1 = *(const float4*)(asrc + jA + 4);
        const float4 s2 = *(const float4*)(asrc + jB);
        const float4 s3 = *(const float4*)(asrc + jB + 4);
        const float sv[16] = {s0.x,s0.y,s0.z,s0.w, s1.x,s1.y,s1.z,s1.w,
                              s2.x,s2.y,s2.z,s2.w, s3.x,s3.y,s3.z,s3.w};

        v16h pa[2];
        #pragma unroll
        for (int g = 0; g < 2; ++g) {
            const float4 g0 = *(const float4*)(adjrowv[g] + jA);
            const float4 g1 = *(const float4*)(adjrowv[g] + jA + 4);
            const float4 g2 = *(const float4*)(adjrowv[g] + jB);
            const float4 g3 = *(const float4*)(adjrowv[g] + jB + 4);
            const float gv[16] = {g0.x,g0.y,g0.z,g0.w, g1.x,g1.y,g1.z,g1.w,
                                  g2.x,g2.y,g2.z,g2.w, g3.x,g3.y,g3.z,g3.w};
            float psum = 0.f;
            #pragma unroll
            for (int e = 0; e < 16; ++e) {
                int   j = (e < 8) ? (jA + e) : (jB + e - 8);
                float a = sv[e] + adstv[g];
                a = (a > 0.f) ? a : NEG_SLOPE * a;          // leaky_relu
                bool  o = (gv[e] != 0.f) || (j == iR[g]);   // adjacency+selfloop
                float p = o ? __expf(a - boundv[g]) : 0.f;
                psum += p;
                pa[g][e] = (_Float16)p;
            }
            psum += __shfl_xor(psum, 16, 32);               // combine K-halves
            lsum[g] += psum;
        }

        // 4 shared B-tiles feed 8 WMMAs (both row groups)
        #pragma unroll
        for (int t = 0; t < 4; ++t) {
            const _Float16* col = vbase + (size_t)(t * 16 + ml) * N_;
            v8h lo = *(const v8h*)(col + jA);
            v8h hi = *(const v8h*)(col + jB);
            v16h vb;
            #pragma unroll
            for (int e = 0; e < 8; ++e) { vb[e] = lo[e]; vb[e + 8] = hi[e]; }
            acc[0][t] = __builtin_amdgcn_wmma_f32_16x16x32_f16(
                false, pa[0], false, vb, (short)0, acc[0][t], false, false);
            acc[1][t] = __builtin_amdgcn_wmma_f32_16x16x32_f16(
                false, pa[1], false, vb, (short)0, acc[1][t], false, false);
        }
    }

    // Normalize by softmax denominator, add bias, store f32 output [B,N,H*C]
    #pragma unroll
    for (int g = 0; g < 2; ++g) {
        const float linv = 1.0f / lsum[g];
        #pragma unroll
        for (int r = 0; r < 8; ++r) {
            float fac = __shfl(linv, rbase + r, 32);
            int   row = i0 + g * 16 + rbase + r;
            float* orow = out + ((size_t)b * N_ + row) * HC_ + h * C_;
            #pragma unroll
            for (int t = 0; t < 4; ++t) {
                int c = t * 16 + ml;
                orow[c] = acc[g][t][r] * fac + bias[h * C_ + c];
            }
        }
    }
}

// ---------------------------------------------------------------------------
// Host launcher
// ---------------------------------------------------------------------------
extern "C" void kernel_launch(void* const* d_in, const int* in_sizes, int n_in,
                              void* d_out, int out_size, void* d_ws, size_t ws_size,
                              hipStream_t stream) {
    const float* x       = (const float*)d_in[0];   // [B,N,F]
    const float* adj     = (const float*)d_in[1];   // [B,N,N]
    const float* W       = (const float*)d_in[2];   // [F,H*C]
    const float* att_src = (const float*)d_in[3];   // [H,C]
    const float* att_dst = (const float*)d_in[4];   // [H,C]
    const float* bias    = (const float*)d_in[5];   // [H*C]
    float*       out     = (float*)d_out;           // [B,N,H*C]

    char* ws = (char*)d_ws;
    const size_t WT_BYTES   = (size_t)HC_ * F_ * sizeof(_Float16);          // 64 KB
    const size_t XHT_BYTES  = (size_t)B_ * H_ * C_ * N_ * sizeof(_Float16); // 4 MB
    const size_t AVEC_BYTES = (size_t)B_ * H_ * N_ * sizeof(float);         // 128 KB

    _Float16* Wt    = (_Float16*)(ws);
    _Float16* xh_t  = (_Float16*)(ws + WT_BYTES);
    float*    a_src = (float*)(ws + WT_BYTES + XHT_BYTES);
    float*    a_dst = (float*)(ws + WT_BYTES + XHT_BYTES + AVEC_BYTES);
    float*    amax  = (float*)(ws + WT_BYTES + XHT_BYTES + 2 * AVEC_BYTES);

    // 1) transpose + convert W
    prep_w_kernel<<<(F_ * HC_ + 255) / 256, 256, 0, stream>>>(W, Wt);

    // 2) projection GEMM + attention coefficients (512 blocks x 4 waves)
    proj_kernel<<<(B_ * N_) / 16, 128, 0, stream>>>(
        x, att_src, att_dst, Wt, xh_t, a_src, a_dst);

    // 3) per-(b,h) max of a_src -> softmax upper bound
    amax_kernel<<<B_ * H_, 256, 0, stream>>>(a_src, amax);

    // 4) fused masked softmax + aggregation (1024 32-row tiles, 4 waves/block)
    gat_attn_kernel<<<(B_ * H_ * (N_ / 32)) / 4, 128, 0, stream>>>(
        adj, a_src, a_dst, amax, xh_t, bias, out);
}
